// GPT2Attention_4337916970020
// MI455X (gfx1250) — matrix operand order repaired and checked by reading, not compiled, not run on hardware
//
#include <hip/hip_runtime.h>
#include <hip/hip_bf16.h>

typedef _Float16 half_t;
typedef __attribute__((ext_vector_type(16))) _Float16 v16h;
typedef __attribute__((ext_vector_type(2)))  _Float16 v2h;
typedef __attribute__((ext_vector_type(8)))  float    v8f;

// Types for the async-to-LDS builtin (param type per hipcc diagnostic:
// 'int __attribute__((vector_size(16))) __device__ *').
typedef int v4i __attribute__((vector_size(16)));
typedef __attribute__((address_space(1))) v4i gv4i;
typedef __attribute__((address_space(3))) v4i lv4i;

#define N_EMBD   768
#define N_HEAD   12
#define HEAD_DIM 64
#define BATCH    2
#define SEQ      2048
#define QKV_N    (3 * N_EMBD)     // 2304
#define MROWS    (BATCH * SEQ)    // 4096

__device__ __forceinline__ v8f wmma_f16(v16h a, v16h b, v8f c) {
  // D = A(16x32 f16) x B(32x16 f16) + C(16x16 f32)
  return __builtin_amdgcn_wmma_f32_16x16x32_f16(
      /*neg_a=*/false, a, /*neg_b=*/false, b,
      /*c_mod=*/(short)0, c, /*reuse_a=*/false, /*reuse_b=*/false);
}

// A fragment: 16x32 tile, row-major half buffer with leading dim ld.
// Lane group 0 (lanes 0-15) holds K=0..15; group 1 holds K=16..31.
// Per-lane halves are two contiguous 8-half runs -> 2x (ds|global)_load_b128.
__device__ __forceinline__ v16h frag_a(const half_t* base, int ld) {
  const int lane = threadIdx.x & 31;
  const int l16 = lane & 15, g = lane >> 4;
  const half_t* row = base + l16 * ld;
  v16h a;
#pragma unroll
  for (int i = 0; i < 8; ++i) {
    a[i]     = row[g * 16 + i];       // K = g*16 + 0..7
    a[8 + i] = row[g * 16 + 8 + i];   // K = g*16 + 8..15
  }
  return a;
}

// B fragment read from frag-ordered LDS: one aligned 32B per-lane load.
__device__ __forceinline__ v16h frag_b_lds(const half_t* sub_base) {
  const int lane = threadIdx.x & 31;
  return *(const v16h*)(sub_base + lane * 16);
}

// Position of element (k,n) of a 32xN B tile in frag order:
// [n>>4][ (k>>4)*16 + (n&15) ][ k&15 ]  with inner dims 32x16.
__device__ __forceinline__ int bfrag_pos(int k, int n) {
  return ((n >> 4) * 32 + (k >> 4) * 16 + (n & 15)) * 16 + (k & 15);
}

// Stage a 32x64 f32 B tile into frag-ordered f16 LDS, 256 threads.
// Each thread: one k-row pair x 4 columns -> 4x v_cvt_pk_f16_f32 + b32 LDS stores.
__device__ __forceinline__ void stage_b_tile(half_t* Bsf, const float* Wrow0,
                                             int ldw, int cb, int k) {
  const float* w0 = Wrow0;
  const float* w1 = Wrow0 + ldw;
#pragma unroll
  for (int j = 0; j < 4; ++j) {
    v2h pair = {(half_t)w0[j], (half_t)w1[j]};
    *(v2h*)&Bsf[bfrag_pos(k, cb + j)] = pair;   // (k&15) even -> 4B aligned
  }
}

// ---------------- Kernel 1: QKV = X @ W_attn + b ----------------------------
// Writes Q,K head-split [B,H,S,64] f16; V transposed [B,H,64,S] f16.
__global__ __launch_bounds__(256)
void qkv_kernel(const float* __restrict__ X, const float* __restrict__ W,
                const float* __restrict__ bias, half_t* __restrict__ Qh,
                half_t* __restrict__ Kh, half_t* __restrict__ Vt) {
  __shared__ __align__(32) half_t As[128][32];      // 8 KB
  __shared__ __align__(32) half_t Bsf[4 * 32 * 16]; // 4 KB, frag-ordered B tile
  const int tid = threadIdx.x;
  const int m0 = blockIdx.y * 128;
  const int n0 = blockIdx.x * 64;
  const int wave = tid >> 5;
  const int wm = wave & 3, wn = wave >> 2;

  v8f acc[2][2] = {};
  const int arow = tid >> 1, acb = (tid & 1) * 16;
  const int bcb = (tid & 15) * 4, bk = (tid >> 4) * 2;

  for (int k0 = 0; k0 < N_EMBD; k0 += 32) {
    __syncthreads();
    const float* xsrc = X + (size_t)(m0 + arow) * N_EMBD + k0 + acb;
#pragma unroll
    for (int j = 0; j < 16; ++j) As[arow][acb + j] = (half_t)xsrc[j];
    const float* wsrc = W + (size_t)(k0 + bk) * QKV_N + n0 + bcb;
    stage_b_tile(Bsf, wsrc, QKV_N, bcb, bk);
    if (k0 + 32 < N_EMBD) {
      __builtin_prefetch(xsrc + 32, 0, 0);                 // global_prefetch_b8
      __builtin_prefetch(wsrc + (size_t)32 * QKV_N, 0, 0);
    }
    __syncthreads();
#pragma unroll
    for (int i = 0; i < 2; ++i) {
      v16h a = frag_a(&As[wm * 32 + i * 16][0], 32);
#pragma unroll
      for (int j = 0; j < 2; ++j) {
        v16h b = frag_b_lds(&Bsf[(wn * 2 + j) * 32 * 16]);
        acc[i][j] = wmma_f16(a, b, acc[i][j]);
      }
    }
  }

  // Epilogue: affine base + constant-offset stores.  which/h/eb are
  // wave-scalar; per-lane part is only +l16 / +8g; per-r offsets constant.
  const int lane = tid & 31, l16 = lane & 15, g = lane >> 4;
  const int which = n0 / N_EMBD;     // block-uniform: Q / K / V
  const int d0 = n0 - which * N_EMBD;
  const int bI = m0 >> 11;           // block-uniform (128 | 2048)
  const int sb = m0 & 2047;
  if (which == 2) {
    // V^T: [B,H,64,S]
#pragma unroll
    for (int i = 0; i < 2; ++i)
#pragma unroll
      for (int j = 0; j < 2; ++j) {
        const int dj = d0 + wn * 32 + j * 16;       // scalar; chunk within head
        const int h = dj >> 6, eb = dj & 63;
        const int sr = sb + wm * 32 + i * 16 + 8 * g;
        const float bn = bias[n0 + wn * 32 + j * 16 + l16];
        half_t* dst =
            Vt + (((size_t)bI * N_HEAD + h) * HEAD_DIM + eb + l16) * SEQ + sr;
#pragma unroll
        for (int r = 0; r < 8; ++r) dst[r] = (half_t)(acc[i][j][r] + bn);
      }
  } else {
    half_t* QK = which ? Kh : Qh;    // [B,H,S,64]
#pragma unroll
    for (int i = 0; i < 2; ++i)
#pragma unroll
      for (int j = 0; j < 2; ++j) {
        const int dj = d0 + wn * 32 + j * 16;
        const int h = dj >> 6, eb = dj & 63;
        const int sr = sb + wm * 32 + i * 16 + 8 * g;
        const float bn = bias[n0 + wn * 32 + j * 16 + l16];
        half_t* dst =
            QK + (((size_t)bI * N_HEAD + h) * SEQ + sr) * HEAD_DIM + eb + l16;
#pragma unroll
        for (int r = 0; r < 8; ++r)
          dst[r * HEAD_DIM] = (half_t)(acc[i][j][r] + bn);
      }
  }
}

// ---------------- Kernel 2: causal flash attention per (b,h) -----------------
__global__ __launch_bounds__(256)
void attn_kernel(const half_t* __restrict__ Qh, const half_t* __restrict__ Kh,
                 const half_t* __restrict__ Vt, half_t* __restrict__ Oh) {
  __shared__ __align__(32) half_t Ps[8][16][32];  // 8 KB: per-wave P (C->A relayout)
  const int tid = threadIdx.x;
  const int wave = tid >> 5;
  const int lane = tid & 31, l16 = lane & 15, g = lane >> 4;
  const int bh = blockIdx.x >> 4;
  const int qchunk = blockIdx.x & 15;
  const int q0 = qchunk * 128 + wave * 16;  // 16 query rows per wave

  const half_t* Qb = Qh + (size_t)bh * SEQ * HEAD_DIM;
  const half_t* Kb = Kh + (size_t)bh * SEQ * HEAD_DIM;
  const half_t* Vb = Vt + (size_t)bh * HEAD_DIM * SEQ;   // [64][2048]
  half_t* Ob = Oh + (size_t)bh * SEQ * HEAD_DIM;

  // Q A-fragments for hd 0..31 / 32..63 (contraction dim of the score GEMM)
  v16h qa0, qa1;
  {
    const half_t* qrow = Qb + (size_t)(q0 + l16) * HEAD_DIM;
#pragma unroll
    for (int i = 0; i < 8; ++i) {
      qa0[i]     = qrow[g * 16 + i];
      qa0[8 + i] = qrow[g * 16 + 8 + i];
      qa1[i]     = qrow[32 + g * 16 + i];
      qa1[8 + i] = qrow[32 + g * 16 + 8 + i];
    }
  }

  v8f o[4] = {};
  float m_run[8], l_run[8];
#pragma unroll
  for (int r = 0; r < 8; ++r) { m_run[r] = -3.0e38f; l_run[r] = 0.0f; }

  for (int kb = 0; kb <= q0 + 15; kb += 32) {   // wave-uniform trip count
    // Scores: S[16x32] = Q(16x64) @ K^T(64x32); K^T B-frags are contiguous
    // 32B per lane (lane = key column) -> global_load_b128 pairs.
    v8f s0 = {}, s1 = {};
    {
      const half_t* k0r = Kb + (size_t)(kb + l16) * HEAD_DIM;
      const half_t* k1r = Kb + (size_t)(kb + 16 + l16) * HEAD_DIM;
      v16h b00 = *(const v16h*)(k0r + g * 16);        // hd 0..31 part
      v16h b01 = *(const v16h*)(k0r + 32 + g * 16);   // hd 32..63 part
      v16h b10 = *(const v16h*)(k1r + g * 16);
      v16h b11 = *(const v16h*)(k1r + 32 + g * 16);
      s0 = wmma_f16(qa0, b00, s0);
      s0 = wmma_f16(qa1, b01, s0);
      s1 = wmma_f16(qa0, b10, s1);
      s1 = wmma_f16(qa1, b11, s1);
    }

    // Scale, causal mask, online softmax. Row = r + 8*g in C layout.
    const float scale = 0.125f;  // 1/sqrt(64)
    float p0[8], p1[8], corr[8], bmax[8];
#pragma unroll
    for (int r = 0; r < 8; ++r) {
      int row = q0 + r + 8 * g;
      float v0 = s0[r] * scale;
      float v1 = s1[r] * scale;
      if (kb + l16 > row)      v0 = -3.0e38f;
      if (kb + 16 + l16 > row) v1 = -3.0e38f;
      s0[r] = v0; s1[r] = v1;
      float bm = fmaxf(v0, v1);
#pragma unroll
      for (int off = 1; off < 16; off <<= 1)
        bm = fmaxf(bm, __shfl_xor(bm, off, 32));
      bmax[r] = bm;
    }
#pragma unroll
    for (int r = 0; r < 8; ++r) {
      float mn = fmaxf(m_run[r], bmax[r]);
      corr[r] = __expf(m_run[r] - mn);
      m_run[r] = mn;
      p0[r] = __expf(s0[r] - mn);
      p1[r] = __expf(s1[r] - mn);
      float rs = p0[r] + p1[r];
#pragma unroll
      for (int off = 1; off < 16; off <<= 1)
        rs += __shfl_xor(rs, off, 32);
      l_run[r] = l_run[r] * corr[r] + rs;
      Ps[wave][r + 8 * g][l16]      = (half_t)p0[r];
      Ps[wave][r + 8 * g][16 + l16] = (half_t)p1[r];
    }
#pragma unroll
    for (int c = 0; c < 4; ++c)
#pragma unroll
      for (int r = 0; r < 8; ++r) o[c][r] *= corr[r];

    // Wave-local LDS ordering (no block barrier: waves have different trip counts)
    asm volatile("s_wait_dscnt 0" ::: "memory");

    // O += P(16x32) @ V(32x64).  V is stored transposed [64][S], so the
    // B-fragment is a contiguous 32B per-lane global load.
    v16h pa = frag_a(&Ps[wave][0][0], 32);
#pragma unroll
    for (int c = 0; c < 4; ++c) {
      const half_t* vtrow = Vb + (size_t)(c * 16 + l16) * SEQ + kb + g * 16;
      v16h vb = *(const v16h*)vtrow;
      o[c] = wmma_f16(pa, vb, o[c]);
    }
  }

#pragma unroll
  for (int c = 0; c < 4; ++c)
#pragma unroll
    for (int r = 0; r < 8; ++r) {
      int row = q0 + r + 8 * g;
      int col = c * 16 + l16;
      Ob[(size_t)row * HEAD_DIM + col] = (half_t)(o[c][r] / l_run[r]);
    }
}

// ---------------- Kernel 3: Y = merge_heads(O) @ W_proj + b (f32 out) --------
__global__ __launch_bounds__(256)
void proj_kernel(const half_t* __restrict__ Oh, const float* __restrict__ W,
                 const float* __restrict__ bias, float* __restrict__ Y) {
  __shared__ __align__(32) half_t As[128][32];
  __shared__ __align__(32) half_t Bsf[4 * 32 * 16];
  const int tid = threadIdx.x;
  const int m0 = blockIdx.y * 128;
  const int n0 = blockIdx.x * 64;
  const int wave = tid >> 5;
  const int wm = wave & 3, wn = wave >> 2;

  v8f acc[2][2] = {};
  const int arow = tid >> 1, acb = (tid & 1) * 16;
  const int bcb = (tid & 15) * 4, bk = (tid >> 4) * 2;
  const int m = m0 + arow;
  const int bI = m >> 11, s = m & 2047;

  for (int k0 = 0; k0 < N_EMBD; k0 += 32) {
    __syncthreads();
    {
      int kk = k0 + acb;           // 16-chunk stays inside one head (hd=64)
      int h = kk >> 6, e = kk & 63;
      const half_t* src = Oh + (((size_t)bI * N_HEAD + h) * SEQ + s) * HEAD_DIM + e;
      half_t* dst = &As[arow][acb];
#if defined(__gfx1250__) && __has_builtin(__builtin_amdgcn_global_load_async_to_lds_b128)
      // Raw f16 copy -> async DMA into LDS (ASYNCcnt-tracked), 2x 16B per thread.
      __builtin_amdgcn_global_load_async_to_lds_b128(
          (gv4i*)(v4i*)src, (lv4i*)(v4i*)dst, 0, 0);
      __builtin_amdgcn_global_load_async_to_lds_b128(
          (gv4i*)(v4i*)(src + 8), (lv4i*)(v4i*)(dst + 8), 0, 0);
#else
#pragma unroll
      for (int j = 0; j < 16; ++j) dst[j] = src[j];
#endif
    }
    const float* wsrc = W + (size_t)(k0 + bk) * N_EMBD + n0 + bcb;
    stage_b_tile(Bsf, wsrc, N_EMBD, bcb, bk);
    if (k0 + 32 < N_EMBD) __builtin_prefetch(wsrc + (size_t)32 * N_EMBD, 0, 0);
#if defined(__gfx1250__) && __has_builtin(__builtin_amdgcn_global_load_async_to_lds_b128)
#if __has_builtin(__builtin_amdgcn_s_wait_asynccnt)
    __builtin_amdgcn_s_wait_asynccnt(0);
#else
    asm volatile("s_wait_asynccnt 0" ::: "memory");
#endif
#endif
    __syncthreads();
#pragma unroll
    for (int i = 0; i < 2; ++i) {
      v16h a = frag_a(&As[wm * 32 + i * 16][0], 32);
#pragma unroll
      for (int j = 0; j < 2; ++j) {
        v16h b = frag_b_lds(&Bsf[(wn * 2 + j) * 32 * 16]);
        acc[i][j] = wmma_f16(a, b, acc[i][j]);
      }
    }
  }

  const int lane = tid & 31, l16 = lane & 15, g = lane >> 4;
#pragma unroll
  for (int i = 0; i < 2; ++i)
#pragma unroll
    for (int j = 0; j < 2; ++j) {
      const int mmb = m0 + wm * 32 + i * 16 + 8 * g;
      const int nn = n0 + wn * 32 + j * 16 + l16;
      const float bn = bias[nn];
      float* dst = Y + (size_t)mmb * N_EMBD + nn;
#pragma unroll
      for (int r = 0; r < 8; ++r)
        dst[r * N_EMBD] = acc[i][j][r] + bn;
    }
}

extern "C" void kernel_launch(void* const* d_in, const int* in_sizes, int n_in,
                              void* d_out, int out_size, void* d_ws, size_t ws_size,
                              hipStream_t stream) {
  const float* X  = (const float*)d_in[0];  // [2,2048,768]
  const float* Wa = (const float*)d_in[1];  // [768,2304]
  const float* ba = (const float*)d_in[2];  // [2304]
  const float* Wp = (const float*)d_in[3];  // [768,768]
  const float* bp = (const float*)d_in[4];  // [768]

  const size_t elems = (size_t)BATCH * N_HEAD * SEQ * HEAD_DIM;  // 3,145,728
  half_t* Qh = (half_t*)d_ws;
  half_t* Kh = Qh + elems;
  half_t* Vt = Kh + elems;   // transposed V: [B,H,64,S]
  half_t* Oh = Vt + elems;   // total ws use: 4 * 6 MB = 24 MB

  qkv_kernel<<<dim3(QKV_N / 64, MROWS / 128), 256, 0, stream>>>(X, Wa, ba, Qh, Kh, Vt);
  attn_kernel<<<dim3(BATCH * N_HEAD * (SEQ / 128)), 256, 0, stream>>>(Qh, Kh, Vt, Oh);
  proj_kernel<<<dim3(N_EMBD / 64, MROWS / 128), 256, 0, stream>>>(Oh, Wp, bp, (float*)d_out);
}
